// GINNodeLevel_87729001988302
// MI455X (gfx1250) — compile-verified, matrix operand order checked by reading, not run on hardware
//
#include <hip/hip_runtime.h>

#define DN 128
#define LL 3
#define PP 7
#define BB 64

typedef __attribute__((ext_vector_type(16))) __bf16 v16bf;
typedef __attribute__((ext_vector_type(8)))  __bf16 v8bf;
typedef __attribute__((ext_vector_type(8)))  float  v8f;

// ---------------------------------------------------------------------------
// Per-graph stats conditioning: cond[l][b][d] = statB[l][d] +
//   sum_p nan_to_num(graph_stats[b][p]) * statW[l][p][d]
// ---------------------------------------------------------------------------
__global__ __launch_bounds__(256) void cond_kernel(
    const float* __restrict__ gstats,   // [B,P]
    const float* __restrict__ statWs,   // [L,P,D]
    const float* __restrict__ statBs,   // [L,D]
    float* __restrict__ cond)           // [L,B,D]
{
    int tid = blockIdx.x * 256 + threadIdx.x;
    if (tid >= LL * BB * DN) return;
    int d = tid & (DN - 1);
    int b = (tid >> 7) & (BB - 1);
    int l = tid >> 13;
    float acc = statBs[l * DN + d];
    #pragma unroll
    for (int p = 0; p < PP; ++p) {
        float s = gstats[b * PP + p];
        s = (s != s) ? -100.0f : s;   // nan_to_num(nan=-100)
        acc += s * statWs[(l * PP + p) * DN + d];
    }
    cond[tid] = acc;
}

// ---------------------------------------------------------------------------
// Edge scatter-add: agg[dst] += h[src].  One thread per (edge, 4-float chunk):
// coalesced float4 gather of the source row, 4 native f32 atomics on the dest.
// ---------------------------------------------------------------------------
__global__ __launch_bounds__(256) void scatter_kernel(
    const float* __restrict__ h,
    const int*   __restrict__ src,
    const int*   __restrict__ dst,
    float* __restrict__ agg,
    long long total)
{
    long long tid = (long long)blockIdx.x * 256 + threadIdx.x;
    if (tid >= total) return;
    long long e = tid >> 5;           // edge index
    int q = (int)(tid & 31) * 4;      // feature chunk
    int s = src[e];
    int d = dst[e];
    float4 v = *(const float4*)(h + (size_t)s * DN + q);
    float* p = agg + (size_t)d * DN + q;
    __hip_atomic_fetch_add(p + 0, v.x, __ATOMIC_RELAXED, __HIP_MEMORY_SCOPE_AGENT);
    __hip_atomic_fetch_add(p + 1, v.y, __ATOMIC_RELAXED, __HIP_MEMORY_SCOPE_AGENT);
    __hip_atomic_fetch_add(p + 2, v.z, __ATOMIC_RELAXED, __HIP_MEMORY_SCOPE_AGENT);
    __hip_atomic_fetch_add(p + 3, v.w, __ATOMIC_RELAXED, __HIP_MEMORY_SCOPE_AGENT);
}

// ---------------------------------------------------------------------------
// Fused [nrows,128] @ [128,128] GEMM via v_wmma_f32_16x16x32_bf16.
//   MODE 1: outb = bf16( BN(leaky((A + A2) @ W + bias)) )   (GIN MLP 1st half)
//   MODE 2: outf = leaky(Abf @ W + bias) + cond[batch[row]] (GIN MLP 2nd half)
//   MODE 3: outf = A @ W + bias                             (final FC)
// Block = 256 threads (8 waves); each wave owns a 16-row strip (N % 16 == 0,
// so the row guard is wave-uniform and EXEC stays all-ones for WMMA).
// Weights staged transposed in LDS as bf16 (32 KB) so each B-fragment is two
// contiguous ds_load_b128 per K-step, matching the ISA 16-bit B layout.
// ---------------------------------------------------------------------------
template <int MODE>
__global__ __launch_bounds__(256) void gemm_fused(
    const float*  __restrict__ A,      // MODE 1,3: fp32 [nrows,128]
    const __bf16* __restrict__ Abf,    // MODE 2:  bf16 [nrows,128]
    const float*  __restrict__ A2,     // MODE 1:  agg addend
    const float*  __restrict__ W,      // [128,128] row-major fp32
    float*  __restrict__ outf,         // MODE 2,3
    __bf16* __restrict__ outb,         // MODE 1
    int nrows,
    const float* __restrict__ bias,    // [128]
    const float* __restrict__ gamma,   // MODE 1
    const float* __restrict__ beta,    // MODE 1
    const float* __restrict__ mean,    // MODE 1
    const float* __restrict__ var,     // MODE 1
    const float* __restrict__ cond,    // MODE 2: [B,128]
    const int*   __restrict__ batch)   // MODE 2: [nrows]
{
    __shared__ __bf16 Wt[DN * DN];     // transposed: Wt[n*128 + k], 32 KB

    const int tid = threadIdx.x;
    // Cooperative load + transpose + bf16 convert of the weight matrix
    // (64 KB fp32, L2-resident across the whole grid).
    for (int idx = tid; idx < DN * DN; idx += 256) {
        int k = idx >> 7;
        int n = idx & (DN - 1);
        Wt[n * DN + k] = (__bf16)W[idx];
    }
    __syncthreads();

    const int wave = tid >> 5;
    const int lane = tid & 31;
    const int r16  = lane & 15;
    const int half = lane >> 4;
    const int row0 = blockIdx.x * 128 + wave * 16;
    if (row0 >= nrows) return;         // wave-uniform

    const int arow = row0 + r16;       // this lane's A-matrix row (M = lane%16)

    // Preload all four A-fragments (full K=128 row).
    // ISA 16-bit A 16x32 layout: lane-half 0 -> K = ks*32 + {0..7, 16..23},
    //                            lane-half 1 -> K = ks*32 + {8..15, 24..31}.
    v16bf afrag[4];
    if constexpr (MODE == 2) {
        const __bf16* ap = Abf + (size_t)arow * DN;
        #pragma unroll
        for (int ks = 0; ks < 4; ++ks) {
            const int b0 = ks * 32 + half * 8;
            v8bf lo = *(const v8bf*)(ap + b0);
            v8bf hi = *(const v8bf*)(ap + b0 + 16);
            #pragma unroll
            for (int j = 0; j < 8; ++j) { afrag[ks][j] = lo[j]; afrag[ks][j + 8] = hi[j]; }
        }
    } else {
        const float* ap = A + (size_t)arow * DN;
        #pragma unroll
        for (int ks = 0; ks < 4; ++ks) {
            const int b0 = ks * 32 + half * 8;
            const int b1 = b0 + 16;
            float4 p0 = *(const float4*)(ap + b0);
            float4 p1 = *(const float4*)(ap + b0 + 4);
            float4 q0 = *(const float4*)(ap + b1);
            float4 q1 = *(const float4*)(ap + b1 + 4);
            if constexpr (MODE == 1) {   // fused GIN aggregation add: h + agg
                const float* ap2 = A2 + (size_t)arow * DN;
                float4 t;
                t = *(const float4*)(ap2 + b0);     p0.x += t.x; p0.y += t.y; p0.z += t.z; p0.w += t.w;
                t = *(const float4*)(ap2 + b0 + 4); p1.x += t.x; p1.y += t.y; p1.z += t.z; p1.w += t.w;
                t = *(const float4*)(ap2 + b1);     q0.x += t.x; q0.y += t.y; q0.z += t.z; q0.w += t.w;
                t = *(const float4*)(ap2 + b1 + 4); q1.x += t.x; q1.y += t.y; q1.z += t.z; q1.w += t.w;
            }
            afrag[ks][0]  = (__bf16)p0.x; afrag[ks][1]  = (__bf16)p0.y;
            afrag[ks][2]  = (__bf16)p0.z; afrag[ks][3]  = (__bf16)p0.w;
            afrag[ks][4]  = (__bf16)p1.x; afrag[ks][5]  = (__bf16)p1.y;
            afrag[ks][6]  = (__bf16)p1.z; afrag[ks][7]  = (__bf16)p1.w;
            afrag[ks][8]  = (__bf16)q0.x; afrag[ks][9]  = (__bf16)q0.y;
            afrag[ks][10] = (__bf16)q0.z; afrag[ks][11] = (__bf16)q0.w;
            afrag[ks][12] = (__bf16)q1.x; afrag[ks][13] = (__bf16)q1.y;
            afrag[ks][14] = (__bf16)q1.z; afrag[ks][15] = (__bf16)q1.w;
        }
    }

    // Hoisted per-row graph ids for the conditioning gather (MODE 2).
    int gid[8];
    if constexpr (MODE == 2) {
        #pragma unroll
        for (int r = 0; r < 8; ++r) gid[r] = batch[row0 + half * 8 + r];
    }

    // 8 column tiles of 16 outputs each (full 128 output features).
    #pragma unroll
    for (int ct = 0; ct < 8; ++ct) {
        const int col = ct * 16 + r16;          // this lane's output column (N)
        const __bf16* wc = Wt + (size_t)col * DN;
        v8f acc = {0.f, 0.f, 0.f, 0.f, 0.f, 0.f, 0.f, 0.f};
        #pragma unroll
        for (int ks = 0; ks < 4; ++ks) {
            v8bf blo = *(const v8bf*)(wc + ks * 32 + half * 8);
            v8bf bhi = *(const v8bf*)(wc + ks * 32 + 16 + half * 8);
            v16bf bfrag;
            #pragma unroll
            for (int j = 0; j < 8; ++j) { bfrag[j] = blo[j]; bfrag[j + 8] = bhi[j]; }
            acc = __builtin_amdgcn_wmma_f32_16x16x32_bf16(
                false, afrag[ks], false, bfrag, (short)0, acc, false, false);
        }

        // Epilogue. C/D layout: lane-half h, VGPR r -> row M = h*8 + r, col = lane%16.
        if constexpr (MODE == 1) {
            const float bc = bias[col];
            const float sc = gamma[col] * rsqrtf(var[col] + 1e-5f);
            const float sh = beta[col] - mean[col] * sc;
            #pragma unroll
            for (int r = 0; r < 8; ++r) {
                const int orow = row0 + half * 8 + r;
                float v = acc[r] + bc;
                v = fmaxf(v, 0.2f * v);                  // LeakyReLU(0.2)
                outb[(size_t)orow * DN + col] = (__bf16)(v * sc + sh);
            }
        } else if constexpr (MODE == 2) {
            const float bc = bias[col];
            #pragma unroll
            for (int r = 0; r < 8; ++r) {
                const int orow = row0 + half * 8 + r;
                float v = acc[r] + bc;
                v = fmaxf(v, 0.2f * v);
                outf[(size_t)orow * DN + col] = v + cond[gid[r] * DN + col];
            }
        } else {
            const float bc = bias[col];
            #pragma unroll
            for (int r = 0; r < 8; ++r) {
                const int orow = row0 + half * 8 + r;
                outf[(size_t)orow * DN + col] = acc[r] + bc;
            }
        }
    }
}

// ---------------------------------------------------------------------------
// Host-side orchestration (graph-capture safe: only async ops on `stream`).
// ---------------------------------------------------------------------------
extern "C" void kernel_launch(void* const* d_in, const int* in_sizes, int n_in,
                              void* d_out, int out_size, void* d_ws, size_t ws_size,
                              hipStream_t stream) {
    const float* x        = (const float*)d_in[0];
    const int*   eidx     = (const int*)  d_in[1];   // [2,E]
    const int*   batch    = (const int*)  d_in[2];   // [N]
    const float* gstats   = (const float*)d_in[3];   // [B,P]
    const float* W1s      = (const float*)d_in[4];   // [L,D,D]
    const float* b1s      = (const float*)d_in[5];
    const float* gammas   = (const float*)d_in[6];
    const float* betas    = (const float*)d_in[7];
    const float* bn_means = (const float*)d_in[8];
    const float* bn_vars  = (const float*)d_in[9];
    const float* W2s      = (const float*)d_in[10];
    const float* b2s      = (const float*)d_in[11];
    const float* statWs   = (const float*)d_in[12];  // [L,P,D]
    const float* statBs   = (const float*)d_in[13];  // [L,D]
    const float* fcW      = (const float*)d_in[14];  // [D,D]
    const float* fcB      = (const float*)d_in[15];  // [D]

    const int       N = in_sizes[0] / DN;
    const long long E = (long long)in_sizes[1] / 2;
    float* out = (float*)d_out;

    char*  ws    = (char*)d_ws;
    size_t nd    = (size_t)N * DN * sizeof(float);
    float* hbuf  = (float*)(ws);                 // [N,D] fp32 layer output
    float* agg   = (float*)(ws + nd);            // [N,D] fp32 neighbor sums
    float* cond  = (float*)(ws + 2 * nd);        // [L,B,D] stats conditioning
    // d_out (N*D fp32 = 2x the bytes needed) doubles as the bf16 X1 scratch
    // between GEMM1 and GEMM2; the final FC overwrites it with the result.
    __bf16* xbuf = (__bf16*)d_out;               // [N,D] bf16 intermediate

    cond_kernel<<<(LL * BB * DN + 255) / 256, 256, 0, stream>>>(
        gstats, statWs, statBs, cond);

    const long long sc_total  = E * 32;
    const int       sc_blocks = (int)((sc_total + 255) / 256);
    const int       gemm_grid = (N + 127) / 128;

    const float* hcur = x;
    for (int i = 0; i < LL; ++i) {
        hipMemsetAsync(agg, 0, nd, stream);
        scatter_kernel<<<sc_blocks, 256, 0, stream>>>(
            hcur, eidx, eidx + E, agg, sc_total);
        // X1 = bf16( BN(leaky((h + agg) @ W1 + b1)) )
        gemm_fused<1><<<gemm_grid, 256, 0, stream>>>(
            hcur, nullptr, agg, W1s + (size_t)i * DN * DN, nullptr, xbuf, N,
            b1s + i * DN, gammas + i * DN, betas + i * DN,
            bn_means + i * DN, bn_vars + i * DN, nullptr, nullptr);
        // h = leaky(X1 @ W2 + b2) + cond[batch]
        gemm_fused<2><<<gemm_grid, 256, 0, stream>>>(
            nullptr, xbuf, nullptr, W2s + (size_t)i * DN * DN, hbuf, nullptr, N,
            b2s + i * DN, nullptr, nullptr, nullptr, nullptr,
            cond + (size_t)i * BB * DN, batch);
        hcur = hbuf;
    }
    // out = h @ fcW + fcB
    gemm_fused<3><<<gemm_grid, 256, 0, stream>>>(
        hbuf, nullptr, nullptr, fcW, out, nullptr, N,
        fcB, nullptr, nullptr, nullptr, nullptr, nullptr, nullptr);
}
